// MessagePassing_55405078118490
// MI455X (gfx1250) — matrix operand order repaired and checked by reading, not compiled, not run on hardware
//
#include <hip/hip_runtime.h>

// ---------------- CDNA5 WMMA types ----------------
typedef __attribute__((ext_vector_type(16))) __bf16 v16bf;
typedef __attribute__((ext_vector_type(8)))  float  v8f;

union ABfrag { int4 q[2]; v16bf v; };

#define BN_EPS 1e-5f

#if __has_builtin(__builtin_amdgcn_global_load_async_to_lds_b128)
#define HAVE_ASYNC_LDS 1
// exact parameter types per hipcc diagnostic:
//   (int __attribute__((vector_size(16))) __device__*, ... __shared__*, imm, imm)
typedef int vint4 __attribute__((vector_size(16)));
typedef __attribute__((address_space(1))) vint4 as1_vint4;
typedef __attribute__((address_space(3))) vint4 as3_vint4;
#else
#define HAVE_ASYNC_LDS 0
#endif

__device__ __forceinline__ unsigned short f2bf(float f) {
  unsigned u = __float_as_uint(f);
  u += 0x7FFFu + ((u >> 16) & 1u);      // round-to-nearest-even
  return (unsigned short)(u >> 16);
}
__device__ __forceinline__ float bf2f(unsigned short h) {
  return __uint_as_float(((unsigned)h) << 16);
}

// ---------------- weight fp32 -> bf16 ----------------
__global__ void cvt_bf16_kernel(const float* __restrict__ w,
                                unsigned short* __restrict__ o, int n) {
  int i = blockIdx.x * blockDim.x + threadIdx.x;
  if (i < n) o[i] = f2bf(w[i]);
}

// ---------------- layer 0: edge(Mx16) @ W0^T(16x512) -> bf16 ----------------
__global__ void l0_gemm_kernel(const float* __restrict__ edge,
                               const float* __restrict__ W0,
                               unsigned short* __restrict__ h, int nEdges) {
  int idx = blockIdx.x * blockDim.x + threadIdx.x;
  if (idx >= nEdges * 512) return;
  int e = idx >> 9, n = idx & 511;
  const float4* er = (const float4*)(edge + (size_t)e * 16);
  const float4* wr = (const float4*)(W0 + (size_t)n * 16);
  float s = 0.f;
#pragma unroll
  for (int q = 0; q < 4; ++q) {
    float4 a = er[q], b = wr[q];
    s += a.x * b.x + a.y * b.y + a.z * b.z + a.w * b.w;
  }
  h[idx] = f2bf(s);
}

// ---------------- per-column sum / sumsq over bf16 matrix ----------------
template <bool NT>
__global__ void col_stats_kernel(const unsigned short* __restrict__ h, int M, int N,
                                 int rowsPerBlock, float* __restrict__ sum,
                                 float* __restrict__ sumsq) {
  int n = blockIdx.x * blockDim.x + threadIdx.x;     // column
  if (n >= N) return;
  int r0 = blockIdx.y * rowsPerBlock;
  int r1 = min(M, r0 + rowsPerBlock);
  float s = 0.f, q = 0.f;
  for (int m = r0; m < r1; ++m) {
    unsigned short raw = NT ? __builtin_nontemporal_load(h + (size_t)m * N + n)
                            : h[(size_t)m * N + n];
    float v = bf2f(raw);
    s += v; q += v * v;
  }
  atomicAdd(&sum[n], s);
  atomicAdd(&sumsq[n], q);
}

// ---------------- fold BN into per-column scale/shift ----------------
__global__ void scale_shift_kernel(const float* __restrict__ sum,
                                   const float* __restrict__ sumsq,
                                   const float* __restrict__ g,
                                   const float* __restrict__ b,
                                   float invM, float* __restrict__ sc,
                                   float* __restrict__ sh, int N) {
  int n = blockIdx.x * blockDim.x + threadIdx.x;
  if (n >= N) return;
  float mean = sum[n] * invM;
  float var  = fmaxf(sumsq[n] * invM - mean * mean, 0.f);
  float s = rsqrtf(var + BN_EPS) * g[n];
  sc[n] = s;
  sh[n] = b[n] - mean * s;
}

// ---------------- in-place BN + ReLU (bf16), N is power of two ----------------
__global__ void bn_relu_kernel(unsigned short* __restrict__ h,
                               const float* __restrict__ sc,
                               const float* __restrict__ sh,
                               size_t total, unsigned nMask) {
  size_t i = (size_t)blockIdx.x * blockDim.x + threadIdx.x;
  if (i >= total) return;
  unsigned n = (unsigned)i & nMask;
  float v = fmaf(bf2f(h[i]), sc[n], sh[n]);
  h[i] = f2bf(fmaxf(v, 0.f));
}

// ---------------- WMMA bf16 GEMM: C(MxN,bf16) = A(MxK,bf16) @ B(NxK,bf16)^T --
// Register-blocked: one wave -> 80x32 super-tile (5 m-tiles x 2 n-tiles).
// Per K-step(32): 14 b128 loads feed 10 v_wmma (B frags reused 5x, A frags 2x).
template <int K, bool NTSTORE>
__global__ void wmma_gemm_bf16(const unsigned short* __restrict__ A,
                               const unsigned short* __restrict__ B,
                               unsigned short* __restrict__ C,
                               int superM, int N) {
  const int lane  = threadIdx.x & 31;
  const int wave  = threadIdx.x >> 5;
  const int stN   = N >> 5;                    // super-tiles along N (32 cols each)
  const int stile = blockIdx.x * 8 + wave;
  if (stile >= superM * stN) return;           // wave-uniform: EXEC stays all-ones
  const int sm = stile / stN;
  const int sn = stile - sm * stN;

  const int mr = lane & 15;                    // row (A) / col (B) within 16-tile
  const int hi = lane >> 4;                    // lane-half selector
  // A 16x32 bf16 frag: kbase 0/8; dwords 0-3 -> K+0..7, dwords 4-7 -> K+16..23
  const unsigned short* arow = A + ((size_t)sm * 80 + mr) * K + hi * 8;
  // B 32x16 bf16 frag: kbase 0/8; dwords 0-3 -> K+0..7, dwords 4-7 -> K+16..23
  const unsigned short* brow = B + ((size_t)sn * 32 + mr) * K + hi * 8;

  v8f acc[5][2] = {};
#pragma unroll 1
  for (int k0 = 0; k0 < K; k0 += 32) {
    ABfrag a[5], b[2];
#pragma unroll
    for (int j = 0; j < 2; ++j) {
      const unsigned short* bp = brow + (size_t)j * 16 * K + k0;
      b[j].q[0] = *(const int4*)(bp);
      b[j].q[1] = *(const int4*)(bp + 16);
    }
#pragma unroll
    for (int i = 0; i < 5; ++i) {
      const unsigned short* ap = arow + (size_t)i * 16 * K + k0;
      a[i].q[0] = *(const int4*)(ap);
      a[i].q[1] = *(const int4*)(ap + 16);
    }
#pragma unroll
    for (int i = 0; i < 5; ++i)
#pragma unroll
      for (int j = 0; j < 2; ++j)
        acc[i][j] = __builtin_amdgcn_wmma_f32_16x16x32_bf16(
            false, a[i].v, false, b[j].v, (short)0, acc[i][j], false, false);
  }
  // C/D layout: VGPR r, lane l -> M = r + 8*(l>>4), N = l&15
#pragma unroll
  for (int i = 0; i < 5; ++i) {
#pragma unroll
    for (int j = 0; j < 2; ++j) {
      unsigned short* crow =
          C + (size_t)(sm * 80 + i * 16 + hi * 8) * N + sn * 32 + j * 16 + mr;
#pragma unroll
      for (int r = 0; r < 8; ++r) {
        unsigned short v = f2bf(acc[i][j][r]);
        if (NTSTORE) __builtin_nontemporal_store(v, crow + (size_t)r * N);
        else         crow[(size_t)r * N] = v;
      }
    }
  }
}

// ---------------- message: BN+ReLU on h3 fused, x_src @ A_e, segment atomics --
// One wave per edge; the 8KB h3 row is staged into LDS with the gfx1250
// async-to-LDS engine (ASYNCcnt) when available, then read via ds_load.
__global__ void __launch_bounds__(128)
msg_kernel(const unsigned short* __restrict__ h3,
           const int* __restrict__ eidx,
           const float* __restrict__ ns,
           const float* __restrict__ sc,
           const float* __restrict__ sh,
           float* __restrict__ summed,
           float* __restrict__ cnt, int nEdges) {
  __shared__ unsigned short hbuf[4 * 4096];    // 4 waves x 8KB
  int wid = (int)((blockIdx.x * (size_t)blockDim.x + threadIdx.x) >> 5);
  if (wid >= nEdges) return;                   // wave-uniform
  int lane = threadIdx.x & 31;
  int wave = threadIdx.x >> 5;
  int src = eidx[(size_t)wid * 2 + 0];
  int dst = eidx[(size_t)wid * 2 + 1];
  const float* x = ns + (size_t)src * 64;
  float xa = x[lane], xb = x[lane + 32];
  const unsigned short* hrow = h3 + (size_t)wid * 4096;

#if HAVE_ASYNC_LDS
  unsigned short* hw = &hbuf[wave * 4096];
#pragma unroll
  for (int i = 0; i < 16; ++i) {
    __builtin_amdgcn_global_load_async_to_lds_b128(
        (as1_vint4*)(hrow + i * 256 + lane * 8),
        (as3_vint4*)(hw + i * 256 + lane * 8),
        0, 0);
  }
#if __has_builtin(__builtin_amdgcn_s_wait_asynccnt)
  __builtin_amdgcn_s_wait_asynccnt(0);
#else
  asm volatile("s_wait_asynccnt 0" ::: "memory");
#endif
  const unsigned short* hsrc = hw;
#else
  const unsigned short* hsrc = hrow;
#endif

  float acc0 = 0.f, acc1 = 0.f;
#pragma unroll 4
  for (int d = 0; d < 64; ++d) {
    float xd = (d < 32) ? __shfl(xa, d, 32) : __shfl(xb, d - 32, 32);
    int j0 = d * 64 + lane, j1 = j0 + 32;
    float a0 = fmaxf(fmaf(bf2f(hsrc[j0]), sc[j0], sh[j0]), 0.f);
    float a1 = fmaxf(fmaf(bf2f(hsrc[j1]), sc[j1], sh[j1]), 0.f);
    acc0 = fmaf(xd, a0, acc0);
    acc1 = fmaf(xd, a1, acc1);
  }
  atomicAdd(&summed[(size_t)dst * 64 + lane], acc0);
  atomicAdd(&summed[(size_t)dst * 64 + lane + 32], acc1);
  if (lane == 0) atomicAdd(&cnt[dst], 1.0f);
}

// ---------------- finalize: mean + bias + relu ----------------
__global__ void finalize_kernel(const float* __restrict__ summed,
                                const float* __restrict__ cnt,
                                const float* __restrict__ bias,
                                float* __restrict__ out, int total) {
  int i = blockIdx.x * blockDim.x + threadIdx.x;
  if (i >= total) return;
  int node = i >> 6, c = i & 63;
  float m = summed[i] / fmaxf(cnt[node], 1.0f);
  out[i] = fmaxf(m + bias[c], 0.f);
}

// =================================================================
extern "C" void kernel_launch(void* const* d_in, const int* in_sizes, int n_in,
                              void* d_out, int out_size, void* d_ws, size_t ws_size,
                              hipStream_t stream) {
  const float* node_states = (const float*)d_in[0];
  const int*   eidx        = (const int*)d_in[1];
  const float* edge        = (const float*)d_in[2];
  const float* W0 = (const float*)d_in[3];
  const float* g0 = (const float*)d_in[4];
  const float* b0 = (const float*)d_in[5];
  const float* W1 = (const float*)d_in[6];
  const float* g1 = (const float*)d_in[7];
  const float* b1 = (const float*)d_in[8];
  const float* W2 = (const float*)d_in[9];
  const float* g2 = (const float*)d_in[10];
  const float* b2 = (const float*)d_in[11];
  const float* bias = (const float*)d_in[12];
  float* out = (float*)d_out;

  const int nNodes = in_sizes[0] / 64;     // 30000
  const int nEdges = in_sizes[2] / 16;     // 60000
  const int M = nEdges;

  // ---- workspace layout ----
  char* base = (char*)d_ws;
  size_t off = 0;
  auto alloc = [&](size_t bytes) -> char* {
    char* p = base + off;
    off += (bytes + 255) & ~(size_t)255;
    return p;
  };
  unsigned short* h1  = (unsigned short*)alloc((size_t)M * 512 * 2);
  unsigned short* h2  = (unsigned short*)alloc((size_t)M * 1024 * 2);
  unsigned short* h3  = (unsigned short*)alloc((size_t)M * 4096 * 2);
  unsigned short* w1b = (unsigned short*)alloc((size_t)1024 * 512 * 2);
  unsigned short* w2b = (unsigned short*)alloc((size_t)4096 * 1024 * 2);
  float* sc0 = (float*)alloc(512 * 4);   float* sh0 = (float*)alloc(512 * 4);
  float* sc1 = (float*)alloc(1024 * 4);  float* sh1 = (float*)alloc(1024 * 4);
  float* sc2 = (float*)alloc(4096 * 4);  float* sh2 = (float*)alloc(4096 * 4);
  // zeroed-every-call region (stats + segment accumulators), kept contiguous
  char* zbase = base + off;
  float* sum0 = (float*)alloc(512 * 4);  float* sq0 = (float*)alloc(512 * 4);
  float* sum1 = (float*)alloc(1024 * 4); float* sq1 = (float*)alloc(1024 * 4);
  float* sum2 = (float*)alloc(4096 * 4); float* sq2 = (float*)alloc(4096 * 4);
  float* summed = (float*)alloc((size_t)nNodes * 64 * 4);
  float* cnt    = (float*)alloc((size_t)nNodes * 4);
  size_t zbytes = (size_t)((base + off) - zbase);
  (void)ws_size; (void)n_in; (void)out_size;

  (void)hipMemsetAsync(zbase, 0, zbytes, stream);

  // weights -> bf16
  cvt_bf16_kernel<<<(1024 * 512 + 255) / 256, 256, 0, stream>>>(W1, w1b, 1024 * 512);
  cvt_bf16_kernel<<<(4096 * 1024 + 255) / 256, 256, 0, stream>>>(W2, w2b, 4096 * 1024);

  const float invM = 1.0f / (float)M;
  const int rowsPB = 512;
  const int rowGrids = (M + rowsPB - 1) / rowsPB;

  // ---- layer 0 ----
  l0_gemm_kernel<<<(M * 512 + 255) / 256, 256, 0, stream>>>(edge, W0, h1, M);
  col_stats_kernel<false><<<dim3(512 / 256, rowGrids), 256, 0, stream>>>(h1, M, 512, rowsPB, sum0, sq0);
  scale_shift_kernel<<<2, 256, 0, stream>>>(sum0, sq0, g0, b0, invM, sc0, sh0, 512);
  bn_relu_kernel<<<(unsigned)(((size_t)M * 512 + 255) / 256), 256, 0, stream>>>(h1, sc0, sh0, (size_t)M * 512, 511u);

  // ---- layer 1: (M x 512) @ (512 x 1024) ----
  {
    int superM = M / 80, stN = 1024 / 32;          // 750 x 32 super-tiles
    int blocks = (superM * stN + 7) / 8;
    wmma_gemm_bf16<512, false><<<blocks, 256, 0, stream>>>(h1, w1b, h2, superM, 1024);
  }
  col_stats_kernel<false><<<dim3(1024 / 256, rowGrids), 256, 0, stream>>>(h2, M, 1024, rowsPB, sum1, sq1);
  scale_shift_kernel<<<4, 256, 0, stream>>>(sum1, sq1, g1, b1, invM, sc1, sh1, 1024);
  bn_relu_kernel<<<(unsigned)(((size_t)M * 1024 + 255) / 256), 256, 0, stream>>>(h2, sc1, sh1, (size_t)M * 1024, 1023u);

  // ---- layer 2: (M x 1024) @ (1024 x 4096), NT stores (h3 >> L2) ----
  {
    int superM = M / 80, stN = 4096 / 32;          // 750 x 128 super-tiles
    int blocks = (superM * stN + 7) / 8;
    wmma_gemm_bf16<1024, true><<<blocks, 256, 0, stream>>>(h2, w2b, h3, superM, 4096);
  }
  col_stats_kernel<true><<<dim3(4096 / 256, rowGrids), 256, 0, stream>>>(h3, M, 4096, rowsPB, sum2, sq2);
  scale_shift_kernel<<<16, 256, 0, stream>>>(sum2, sq2, g2, b2, invM, sc2, sh2, 4096);

  // ---- message passing (layer-2 BN fused) + segment mean ----
  msg_kernel<<<(nEdges + 3) / 4, 128, 0, stream>>>(h3, eidx, node_states, sc2, sh2, summed, cnt, nEdges);
  finalize_kernel<<<(nNodes * 64 + 255) / 256, 256, 0, stream>>>(summed, cnt, bias, out, nNodes * 64);
}
